// Encoder_PN_73194832659141
// MI455X (gfx1250) — matrix operand order repaired and checked by reading, not compiled, not run on hardware
//
#include <hip/hip_runtime.h>
#include <hip/hip_bf16.h>
#include <stdint.h>

typedef __attribute__((ext_vector_type(16))) __bf16 v16bf;
typedef __attribute__((ext_vector_type(8)))  float  v8f;

union FragAB { uint4 q[2]; v16bf v; };

__device__ __forceinline__ unsigned short f2bf(float x) {
    // native v_cvt (RNE) on gfx1250
    return __builtin_bit_cast(unsigned short, static_cast<__bf16>(x));
}

__device__ __forceinline__ v8f wmma_bf16(const FragAB& a, const FragAB& b, v8f c) {
    return __builtin_amdgcn_wmma_f32_16x16x32_bf16(false, a.v, false, b.v,
                                                   (short)0, c, false, false);
}

// A fragment (16-bit 16x32) from row-major bf16 LDS tile, stride in elements (mult of 8).
__device__ __forceinline__ FragAB load_frag_a_lds(const unsigned short* lds, int stride,
                                                  int m0, int k0) {
    int lane = threadIdx.x & 31;
    int base = (m0 + (lane & 15)) * stride + k0 + ((lane >> 4) << 3);
    FragAB f;
    f.q[0] = *reinterpret_cast<const uint4*>(lds + base);
    f.q[1] = *reinterpret_cast<const uint4*>(lds + base + 16);
    return f;
}

// B fragment (16-bit 32x16) from row-major bf16 W[cin][cout]: lane l = column n0+l%16,
// elements in table k order (k0+8h+s, k0+16+8h+s-8), strided scalar reads.
__device__ __forceinline__ FragAB load_frag_b(const unsigned short* __restrict__ W,
                                              int ldw, int k0, int n0) {
    int lane = threadIdx.x & 31;
    const unsigned short* p = W + (size_t)(k0 + ((lane >> 4) << 3)) * ldw + n0 + (lane & 15);
    FragAB f;
#pragma unroll
    for (int s = 0; s < 16; ++s)
        f.v[s] = __builtin_bit_cast(__bf16, p[(size_t)(s < 8 ? s : s + 8) * ldw]);
    return f;
}

// BN+ReLU epilogue -> bf16 store into LDS tile (no accumulator copies).
__device__ __forceinline__ void epilog_store(const v8f& c, int col,
                                             const float* __restrict__ sc,
                                             const float* __restrict__ sh,
                                             unsigned short* buf, int stride,
                                             int m0, int lh) {
    float s = sc[col], t = sh[col];
#pragma unroll
    for (int i = 0; i < 8; ++i) {
        float y = fmaxf(c[i] * s + t, 0.0f);
        buf[(m0 + i + 8 * lh) * stride + col] = f2bf(y);
    }
}

// BN+ReLU epilogue -> per-lane 2-segment max candidates + 2 LDS atomics.
__device__ __forceinline__ void epilog_max(const v8f& c, int col,
                                           const float* __restrict__ sc,
                                           const float* __restrict__ sh,
                                           unsigned segbits, unsigned* lmax, int span) {
    float s = sc[col], t = sh[col];
    float mx0 = 0.f, mx1 = 0.f;
#pragma unroll
    for (int i = 0; i < 8; ++i) {
        float y = fmaxf(c[i] * s + t, 0.0f);
        bool o = (segbits >> i) & 1u;
        mx0 = fmaxf(mx0, o ? 0.f : y);
        mx1 = fmaxf(mx1, o ? y : 0.f);
    }
    atomicMax(&lmax[col],        __builtin_bit_cast(unsigned, mx0));
    atomicMax(&lmax[span + col], __builtin_bit_cast(unsigned, mx1));
}

// k_l12 epilogue: store into staging tile AND feed segment-max.
__device__ __forceinline__ void epilog_store_max(const v8f& c, int col,
                                                 const float* __restrict__ sc,
                                                 const float* __restrict__ sh,
                                                 unsigned short* buf, int stride,
                                                 int m0, int lh,
                                                 unsigned segbits, unsigned* lmax, int span) {
    float s = sc[col], t = sh[col];
    float mx0 = 0.f, mx1 = 0.f;
#pragma unroll
    for (int i = 0; i < 8; ++i) {
        float y = fmaxf(c[i] * s + t, 0.0f);
        buf[(m0 + i + 8 * lh) * stride + col] = f2bf(y);
        bool o = (segbits >> i) & 1u;
        mx0 = fmaxf(mx0, o ? 0.f : y);
        mx1 = fmaxf(mx1, o ? y : 0.f);
    }
    atomicMax(&lmax[col],        __builtin_bit_cast(unsigned, mx0));
    atomicMax(&lmax[span + col], __builtin_bit_cast(unsigned, mx1));
}

// ---------------- prep kernels ----------------
__global__ void k_scale(const float* __restrict__ b, const float* __restrict__ g,
                        const float* __restrict__ be, const float* __restrict__ m,
                        const float* __restrict__ v, float* __restrict__ sc,
                        float* __restrict__ sh, int C) {
    int i = blockIdx.x * blockDim.x + threadIdx.x;
    if (i < C) {
        float s = g[i] * rsqrtf(v[i] + 1e-3f);
        sc[i] = s;
        sh[i] = (b[i] - m[i]) * s + be[i];
    }
}

__global__ void k_cvt(const float* __restrict__ in, unsigned short* __restrict__ out, int n) {
    int i = blockIdx.x * blockDim.x + threadIdx.x;
    if (i < n) out[i] = f2bf(in[i]);
}

__global__ void k_zero(unsigned* __restrict__ p, int n) {
    int i = blockIdx.x * blockDim.x + threadIdx.x;
    if (i < n) p[i] = 0u;
}

// ---------------- fused L1 + L2 + partial segment-max ----------------
__global__ __launch_bounds__(256) void k_l12(
    const float* __restrict__ x, const int* __restrict__ seg_ids,
    const int* __restrict__ nsegp,
    const float* __restrict__ W1, const float* __restrict__ sc1, const float* __restrict__ sh1,
    const unsigned short* __restrict__ w2bf,
    const float* __restrict__ sc2, const float* __restrict__ sh2,
    unsigned short* __restrict__ f2out, unsigned* __restrict__ gmaxU, int N)
{
    // shbuf phase 1: 128x136 bf16 f1 tile (A source); phase 2 (after A frags are
    // VGPR-resident): 128x264 f2 staging tile for coalesced write-back.
    __shared__ __align__(16) unsigned short shbuf[128 * 264];
    __shared__ float    xs[128 * 3];
    __shared__ int      segl[128];
    __shared__ unsigned lmax[2 * 256];
    __shared__ int      seg0s;

    const int tid = threadIdx.x;
    const int R0  = blockIdx.x * 128;

    if (tid < 128) {
        int r = min(R0 + tid, N - 1);
        xs[tid * 3 + 0] = x[r * 3 + 0];
        xs[tid * 3 + 1] = x[r * 3 + 1];
        xs[tid * 3 + 2] = x[r * 3 + 2];
        segl[tid] = seg_ids[r];
    }
    if (tid == 0) seg0s = seg_ids[min(R0, N - 1)];
    for (int j = tid; j < 2 * 256; j += 256) lmax[j] = 0u;
    __syncthreads();

    // Layer 1 (K=3): VALU, BN+ReLU folded, bf16 tile to LDS (stride 136).
    {
        int row = tid >> 1;
        int cb  = (tid & 1) * 64;
        float x0 = xs[row * 3 + 0], x1 = xs[row * 3 + 1], x2 = xs[row * 3 + 2];
        for (int c = cb; c < cb + 64; ++c) {
            float y = x0 * W1[c] + x1 * W1[128 + c] + x2 * W1[256 + c];
            y = fmaxf(y * sc1[c] + sh1[c], 0.0f);
            shbuf[row * 136 + c] = f2bf(y);
        }
    }
    __syncthreads();

    const int w    = tid >> 5;
    const int lane = tid & 31;
    const int lh   = lane >> 4;
    const int ncol = lane & 15;
    const int m0   = w * 16;
    const int seg0 = seg0s;

    // A fragments resident for the whole layer (K=128 -> 4 frags).
    FragAB a0 = load_frag_a_lds(shbuf, 136, m0, 0);
    FragAB a1 = load_frag_a_lds(shbuf, 136, m0, 32);
    FragAB a2 = load_frag_a_lds(shbuf, 136, m0, 64);
    FragAB a3 = load_frag_a_lds(shbuf, 136, m0, 96);
    __syncthreads();   // f1 fully consumed -> shbuf reusable as f2 stage

    unsigned segbits = 0;
#pragma unroll
    for (int i = 0; i < 8; ++i)
        segbits |= (segl[m0 + i + 8 * lh] != seg0) ? (1u << i) : 0u;

    // 16 n-tiles in 4 groups of 4 -> 4 independent accumulator chains.
    for (int ng = 0; ng < 4; ++ng) {
        int n0 = ng * 64;
        v8f c0 = {0.f,0.f,0.f,0.f,0.f,0.f,0.f,0.f};
        v8f c1 = c0, c2 = c0, c3 = c0;
#pragma unroll
        for (int kk = 0; kk < 4; ++kk) {
            int k0 = kk * 32;
            const FragAB& a = (kk == 0) ? a0 : (kk == 1) ? a1 : (kk == 2) ? a2 : a3;
            c0 = wmma_bf16(a, load_frag_b(w2bf, 256, k0, n0 + 0),  c0);
            c1 = wmma_bf16(a, load_frag_b(w2bf, 256, k0, n0 + 16), c1);
            c2 = wmma_bf16(a, load_frag_b(w2bf, 256, k0, n0 + 32), c2);
            c3 = wmma_bf16(a, load_frag_b(w2bf, 256, k0, n0 + 48), c3);
        }
        epilog_store_max(c0, n0 +  0 + ncol, sc2, sh2, shbuf, 264, m0, lh, segbits, lmax, 256);
        epilog_store_max(c1, n0 + 16 + ncol, sc2, sh2, shbuf, 264, m0, lh, segbits, lmax, 256);
        epilog_store_max(c2, n0 + 32 + ncol, sc2, sh2, shbuf, 264, m0, lh, segbits, lmax, 256);
        epilog_store_max(c3, n0 + 48 + ncol, sc2, sh2, shbuf, 264, m0, lh, segbits, lmax, 256);
    }
    __syncthreads();

    // Coalesced f2 write-back: 128 rows x 256 cols bf16 as b128 stores.
#pragma unroll
    for (int it = 0; it < 16; ++it) {
        int j   = tid + 256 * it;          // 4096 uint4 total
        int row = j >> 5;
        int c8  = (j & 31) * 8;
        int rg  = R0 + row;
        if (rg < N) {
            uint4 v = *reinterpret_cast<const uint4*>(&shbuf[row * 264 + c8]);
            *reinterpret_cast<uint4*>(&f2out[(size_t)rg * 256 + c8]) = v;
        }
    }

    const int nseg = nsegp[0];
    for (int j = tid; j < 2 * 256; j += 256) {
        int s = seg0 + (j >> 8);
        if (s < nseg) atomicMax(&gmaxU[s * 256 + (j & 255)], lmax[j]);
    }
}

// ---------------- fused concat + L3 + L4 + final segment-max ----------------
__global__ __launch_bounds__(256) void k_l34(
    const unsigned short* __restrict__ f2in, const int* __restrict__ seg_ids,
    const int* __restrict__ nsegp, const float* __restrict__ gmaxF,
    const unsigned short* __restrict__ w3bf,
    const float* __restrict__ sc3, const float* __restrict__ sh3,
    const unsigned short* __restrict__ w4bf,
    const float* __restrict__ sc4, const float* __restrict__ sh4,
    unsigned* __restrict__ outU, int N)
{
    __shared__ __align__(16) unsigned short a3[64 * 520];
    __shared__ __align__(16) unsigned short h3[64 * 520];
    __shared__ int      segl[64];
    __shared__ unsigned lmax[2 * 1024];
    __shared__ int      seg0s;

    const int tid = threadIdx.x;
    const int R0  = blockIdx.x * 64;

    if (tid < 64) segl[tid] = seg_ids[min(R0 + tid, N - 1)];
    if (tid == 0) seg0s = seg_ids[min(R0, N - 1)];
    for (int j = tid; j < 2 * 1024; j += 256) lmax[j] = 0u;
    __syncthreads();

    // cols [0,256): f2 tile (vectorized bf16 copy)
#pragma unroll
    for (int it = 0; it < 8; ++it) {
        int j   = tid + 256 * it;        // 2048 uint4 total
        int row = j >> 5;
        int c8  = (j & 31) * 8;
        int rg  = min(R0 + row, N - 1);
        uint4 v = *reinterpret_cast<const uint4*>(f2in + (size_t)rg * 256 + c8);
        *reinterpret_cast<uint4*>(&a3[row * 520 + c8]) = v;
    }
    // cols [256,512): gathered global max, f32 -> bf16 (float4 loads, uint2 stores)
    {
        int row = tid >> 2;
        int cb  = (tid & 3) * 64;
        int s   = segl[row];
        const float* gp = gmaxF + s * 256 + cb;
        for (int c4 = 0; c4 < 64; c4 += 4) {
            float4 v = *reinterpret_cast<const float4*>(gp + c4);
            uint2 u;
            u.x = (unsigned)f2bf(v.x) | ((unsigned)f2bf(v.y) << 16);
            u.y = (unsigned)f2bf(v.z) | ((unsigned)f2bf(v.w) << 16);
            *reinterpret_cast<uint2*>(&a3[row * 520 + 256 + cb + c4]) = u;
        }
    }
    __syncthreads();

    const int w    = tid >> 5;
    const int lane = tid & 31;
    const int lh   = lane >> 4;
    const int ncol = lane & 15;
    const int m0   = (w & 3) * 16;           // 4 m-tiles, 2 waves each
    const int seg0 = seg0s;

    unsigned segbits = 0;
#pragma unroll
    for (int i = 0; i < 8; ++i)
        segbits |= (segl[m0 + i + 8 * lh] != seg0) ? (1u << i) : 0u;

    // Layer 3: 16 n-tiles per wave (groups of 4), K=512 (16 WMMA steps).
    const int nbase3 = (w >> 2) * 256;
    for (int grp = 0; grp < 4; ++grp) {
        int n0 = nbase3 + grp * 64;
        v8f c0 = {0.f,0.f,0.f,0.f,0.f,0.f,0.f,0.f};
        v8f c1 = c0, c2 = c0, c3 = c0;
        for (int k = 0; k < 16; ++k) {
            FragAB a = load_frag_a_lds(a3, 520, m0, k * 32);
            c0 = wmma_bf16(a, load_frag_b(w3bf, 512, k * 32, n0 + 0),  c0);
            c1 = wmma_bf16(a, load_frag_b(w3bf, 512, k * 32, n0 + 16), c1);
            c2 = wmma_bf16(a, load_frag_b(w3bf, 512, k * 32, n0 + 32), c2);
            c3 = wmma_bf16(a, load_frag_b(w3bf, 512, k * 32, n0 + 48), c3);
        }
        epilog_store(c0, n0 +  0 + ncol, sc3, sh3, h3, 520, m0, lh);
        epilog_store(c1, n0 + 16 + ncol, sc3, sh3, h3, 520, m0, lh);
        epilog_store(c2, n0 + 32 + ncol, sc3, sh3, h3, 520, m0, lh);
        epilog_store(c3, n0 + 48 + ncol, sc3, sh3, h3, 520, m0, lh);
    }
    __syncthreads();

    // Layer 4: 32 n-tiles per wave (groups of 4), K=512; epilogue feeds the
    // final segment-max (h4 never touches HBM).
    const int nbase4 = (w >> 2) * 512;
    for (int grp = 0; grp < 8; ++grp) {
        int n0 = nbase4 + grp * 64;
        v8f c0 = {0.f,0.f,0.f,0.f,0.f,0.f,0.f,0.f};
        v8f c1 = c0, c2 = c0, c3 = c0;
        for (int k = 0; k < 16; ++k) {
            FragAB a = load_frag_a_lds(h3, 520, m0, k * 32);
            c0 = wmma_bf16(a, load_frag_b(w4bf, 1024, k * 32, n0 + 0),  c0);
            c1 = wmma_bf16(a, load_frag_b(w4bf, 1024, k * 32, n0 + 16), c1);
            c2 = wmma_bf16(a, load_frag_b(w4bf, 1024, k * 32, n0 + 32), c2);
            c3 = wmma_bf16(a, load_frag_b(w4bf, 1024, k * 32, n0 + 48), c3);
        }
        epilog_max(c0, n0 +  0 + ncol, sc4, sh4, segbits, lmax, 1024);
        epilog_max(c1, n0 + 16 + ncol, sc4, sh4, segbits, lmax, 1024);
        epilog_max(c2, n0 + 32 + ncol, sc4, sh4, segbits, lmax, 1024);
        epilog_max(c3, n0 + 48 + ncol, sc4, sh4, segbits, lmax, 1024);
    }
    __syncthreads();

    const int nseg = nsegp[0];
    for (int j = tid; j < 2 * 1024; j += 256) {
        int s = seg0 + (j >> 10);
        if (s < nseg) atomicMax(&outU[s * 1024 + (j & 1023)], lmax[j]);
    }
}

// ---------------- host launcher ----------------
extern "C" void kernel_launch(void* const* d_in, const int* in_sizes, int n_in,
                              void* d_out, int out_size, void* d_ws, size_t ws_size,
                              hipStream_t stream) {
    const float* x       = (const float*)d_in[0];
    const int*   seg_ids = (const int*)d_in[1];
    const int*   nsegp   = (const int*)d_in[2];
    const float* W1 = (const float*)d_in[3];
    const float* W2 = (const float*)d_in[9];
    const float* W3 = (const float*)d_in[15];
    const float* W4 = (const float*)d_in[21];

    const int N = in_sizes[0] / 3;   // 131072

    char* ws = (char*)d_ws;
    float* sc1 = (float*)(ws + 0);   float* sh1 = sc1 + 128;
    float* sc2 = sh1 + 128;          float* sh2 = sc2 + 256;
    float* sc3 = sh2 + 256;          float* sh3 = sc3 + 512;
    float* sc4 = sh3 + 512;          float* sh4 = sc4 + 1024;   // ends @ 15360B
    unsigned*       gmaxU = (unsigned*)(ws + 16384);            // 32*256 u32
    unsigned short* w2bf  = (unsigned short*)(ws + 49152);      // 128*256
    unsigned short* w3bf  = (unsigned short*)(ws + 114688);     // 512*512
    unsigned short* w4bf  = (unsigned short*)(ws + 638976);     // 512*1024
    unsigned short* f2bfv = (unsigned short*)(ws + 1687552);    // N*256 bf16

    k_scale<<<1, 128, 0, stream>>>((const float*)d_in[4],  (const float*)d_in[5],
                                   (const float*)d_in[6],  (const float*)d_in[7],
                                   (const float*)d_in[8],  sc1, sh1, 128);
    k_scale<<<1, 256, 0, stream>>>((const float*)d_in[10], (const float*)d_in[11],
                                   (const float*)d_in[12], (const float*)d_in[13],
                                   (const float*)d_in[14], sc2, sh2, 256);
    k_scale<<<2, 256, 0, stream>>>((const float*)d_in[16], (const float*)d_in[17],
                                   (const float*)d_in[18], (const float*)d_in[19],
                                   (const float*)d_in[20], sc3, sh3, 512);
    k_scale<<<4, 256, 0, stream>>>((const float*)d_in[22], (const float*)d_in[23],
                                   (const float*)d_in[24], (const float*)d_in[25],
                                   (const float*)d_in[26], sc4, sh4, 1024);
    k_cvt<<<(128 * 256 + 255) / 256, 256, 0, stream>>>(W2, w2bf, 128 * 256);
    k_cvt<<<(512 * 512 + 255) / 256, 256, 0, stream>>>(W3, w3bf, 512 * 512);
    k_cvt<<<(512 * 1024 + 255) / 256, 256, 0, stream>>>(W4, w4bf, 512 * 1024);
    k_zero<<<(32 * 256 + 255) / 256, 256, 0, stream>>>(gmaxU, 32 * 256);
    k_zero<<<(out_size + 255) / 256, 256, 0, stream>>>((unsigned*)d_out, out_size);

    k_l12<<<(N + 127) / 128, 256, 0, stream>>>(x, seg_ids, nsegp, W1, sc1, sh1,
                                               w2bf, sc2, sh2, f2bfv, gmaxU, N);
    k_l34<<<(N + 63) / 64, 256, 0, stream>>>(f2bfv, seg_ids, nsegp, (const float*)gmaxU,
                                             w3bf, sc3, sh3, w4bf, sc4, sh4,
                                             (unsigned*)d_out, N);
}
